// ObjectDetr3DCrossAtten_82884278879189
// MI455X (gfx1250) — matrix operand order repaired and checked by reading, not compiled, not run on hardware
//
#include <hip/hip_runtime.h>
#include <hip/hip_bf16.h>

// ---------------------------------------------------------------------------
// MSDeformAttn-style cross attention for MI455X (gfx1250, wave32, WMMA + TDM).
//
// - All GEMMs: v_wmma_f32_16x16x32_bf16 (bf16 in, f32 accumulate).
// - Projected values stored bf16 (89 MB) => L2-resident (192 MB L2) for the
//   bilinear gather stage.
// - Dominant GEMM (value projection, 22.8 GFLOP / 178 MB stream) uses the
//   Tensor Data Mover to DMA A tiles into LDS (TENSORcnt-tracked).
// ---------------------------------------------------------------------------

typedef __attribute__((ext_vector_type(16))) __bf16 v16bf;
typedef __attribute__((ext_vector_type(8)))  float  v8f;
typedef __attribute__((ext_vector_type(4)))  unsigned int u32x4;
typedef __attribute__((ext_vector_type(8)))  unsigned int u32x8;

#define NHD   8
#define NLV   4
#define NPT   4
#define CDIM  256
#define BSZ   8
#define NQRY  900
#define DDIM  32
#define NVAL  21760
#define MQ    (NQRY * BSZ)        // 7200 query rows
#define MV    (NVAL * BSZ)        // 174080 value rows

// ---------------------------------------------------------------------------
// Tensor Data Mover: DMA a 2-D tile of f32 (tile_d0 elems/row x tile_d1 rows)
// from global memory into LDS.  D# built per CDNA5 ISA ch.8:
//   group0: [1:0]=count=1, [63:32]=lds_addr, [120:64]=global_addr, [127:126]=2
//   group1: [17:16]=data_size(2 -> 4B), [79:48]=tensor_dim0, [111:80]=tensor_dim1,
//           [127:112]=tile_dim0, [143:128]=tile_dim1, [207:160]=tensor_dim0_stride
// Inline asm (not the builtin) so the code is portable across toolchain
// builtin-arity differences; completion via s_wait_tensorcnt.
// ---------------------------------------------------------------------------
__device__ __forceinline__ void tdm_load_tile_2d_f32(unsigned ldsoff, const void* gptr,
                                                     unsigned tile_d0, unsigned tile_d1,
                                                     unsigned tensor_d0, unsigned tensor_d1,
                                                     unsigned long long stride0) {
  unsigned long long ga = (unsigned long long)(uintptr_t)gptr;
  u32x4 g0;
  g0[0] = 1u;                                                  // count=1, user desc
  g0[1] = ldsoff;                                              // lds_addr (bytes)
  g0[2] = (unsigned)(ga & 0xFFFFFFFFu);                        // global_addr[31:0]
  g0[3] = (unsigned)((ga >> 32) & 0x01FFFFFFu) | (2u << 30);   // global_addr[56:32], type=2
  u32x8 g1;
  g1[0] = (2u << 16);                                          // wg_mask=0, data_size=4B
  g1[1] = (tensor_d0 & 0xFFFFu) << 16;                         // tensor_dim0[15:0]
  g1[2] = (tensor_d0 >> 16) | ((tensor_d1 & 0xFFFFu) << 16);   // dim0 hi | dim1 lo
  g1[3] = (tensor_d1 >> 16) | ((tile_d0 & 0xFFFFu) << 16);     // dim1 hi | tile_dim0
  g1[4] = (tile_d1 & 0xFFFFu);                                 // tile_dim1 | tile_dim2=0
  g1[5] = (unsigned)(stride0 & 0xFFFFFFFFu);                   // dim0_stride[31:0]
  g1[6] = (unsigned)((stride0 >> 32) & 0xFFFFu);               // dim0_stride[47:32]
  g1[7] = 0u;                                                  // dim1_stride unused (2D)
  asm volatile("tensor_load_to_lds %0, %1" :: "s"(g0), "s"(g1) : "memory");
}

// ---------------------------------------------------------------------------
// B-operand packing: CDNA5 WMMA B matrix (32x16 bf16) lane layout:
//   lanes 0-15  : col N = lane,    elems 0..7 -> K=0..7,  elems 8..15 -> K=16..23
//   lanes 16-31 : col N = lane-16, elems 0..7 -> K=8..15, elems 8..15 -> K=24..31
// ---------------------------------------------------------------------------
__global__ void pack_b_kernel(const float* __restrict__ W, __bf16* __restrict__ Wpk,
                              int K, int N) {
  int Nt = N >> 4;
  int gid = blockIdx.x * blockDim.x + threadIdx.x;
  int lane = gid & 31;
  int tile = gid >> 5;
  if (tile >= (K >> 5) * Nt) return;
  int kt = tile / Nt;
  int nt = tile % Nt;
  int col  = nt * 16 + (lane & 15);
  int krow = kt * 32 + (lane >> 4) * 8;
  __bf16* dst = Wpk + (size_t)tile * 512 + lane * 16;
#pragma unroll
  for (int e = 0; e < 16; ++e)
    dst[e] = (__bf16)W[(size_t)(krow + (e & 7) + ((e >> 3) << 4)) * N + col];
}

// ---------------------------------------------------------------------------
// A fragment (16x32 bf16) lane layout:
//   lanes 0-15  : row M = lane,    elems 0..7 -> K=0..7,   elems 8..15 -> K=16..23
//   lanes 16-31 : row M = lane-16, elems 0..7 -> K=8..15,  elems 8..15 -> K=24..31
// ---------------------------------------------------------------------------
__device__ __forceinline__ v16bf a_frag_from_lds_f32(const float* lds_a, int lane) {
  const float* p = lds_a + (lane & 15) * 32 + ((lane >> 4) * 8);
  v16bf a;
#pragma unroll
  for (int e = 0; e < 8; ++e) a[e]     = (__bf16)p[e];
#pragma unroll
  for (int e = 0; e < 8; ++e) a[e + 8] = (__bf16)p[16 + e];
  return a;
}

__device__ __forceinline__ v16bf a_frag_from_lds_bf16(const __bf16* lds_a, int lane) {
  const __bf16* p = lds_a + (lane & 15) * 32 + ((lane >> 4) * 8);
  v16bf a;
#pragma unroll
  for (int e = 0; e < 8; ++e) a[e]     = p[e];
#pragma unroll
  for (int e = 0; e < 8; ++e) a[e + 8] = p[16 + e];
  return a;
}

__device__ __forceinline__ v16bf a_frag_from_global_bf16(const __bf16* __restrict__ A,
                                                         int lda, int lane) {
  const __bf16* p = A + (size_t)(lane & 15) * lda + ((lane >> 4) * 8);
  v16bf a;
#pragma unroll
  for (int e = 0; e < 8; ++e) a[e]     = p[e];
#pragma unroll
  for (int e = 0; e < 8; ++e) a[e + 8] = p[16 + e];
  return a;
}

__device__ __forceinline__ v16bf b_frag(const __bf16* __restrict__ Wpk, int tile, int lane) {
  return *(const v16bf*)(Wpk + (size_t)tile * 512 + lane * 16);
}

// ---------------------------------------------------------------------------
// Value projection: (MV x 256) f32 @ packed W (256x256) -> bf16 (MV x 256).
// 1 block = 32 output rows; 4 waves x 64 cols, 2 M-subtiles each (8 WMMA/kt).
// A tile (32x32 f32, 4 KB) DMA'd into LDS by the Tensor Data Mover.
// ---------------------------------------------------------------------------
__global__ __launch_bounds__(128)
void gemm_value_kernel(const float* __restrict__ value, const float* __restrict__ b_value,
                       const __bf16* __restrict__ Wpk, __bf16* __restrict__ Vout) {
  __shared__ float lds_a[32 * 32];
  const int mtile = blockIdx.x;          // 32 rows per block
  const int wave  = threadIdx.x >> 5;
  const int lane  = threadIdx.x & 31;
  const float* Abase = value + (size_t)mtile * 32 * CDIM;
  const unsigned ldsoff = (unsigned)(uintptr_t)(&lds_a[0]);

  v8f acc[2][4] = {};
  for (int kt = 0; kt < 8; ++kt) {
    __syncthreads();                     // previous-iteration frag reads done
    if (wave == 0) {
      // DMA 32 rows x 32 f32 (K-slab kt) of the A panel into LDS.
      tdm_load_tile_2d_f32(ldsoff, Abase + kt * 32,
                           /*tile_d0=*/32, /*tile_d1=*/32,
                           /*tensor_d0=*/CDIM, /*tensor_d1=*/(unsigned)(MV * 2),
                           /*stride0=*/(unsigned long long)CDIM);
      __builtin_amdgcn_s_wait_tensorcnt(0);
    }
    __syncthreads();                     // publish tile to all waves

    v16bf a0 = a_frag_from_lds_f32(lds_a, lane);
    v16bf a1 = a_frag_from_lds_f32(lds_a + 16 * 32, lane);
#pragma unroll
    for (int j = 0; j < 4; ++j) {
      v16bf b = b_frag(Wpk, kt * 16 + wave * 4 + j, lane);
      acc[0][j] = __builtin_amdgcn_wmma_f32_16x16x32_bf16(false, a0, false, b,
                                                          (short)0, acc[0][j], false, false);
      acc[1][j] = __builtin_amdgcn_wmma_f32_16x16x32_bf16(false, a1, false, b,
                                                          (short)0, acc[1][j], false, false);
    }
  }
#pragma unroll
  for (int m = 0; m < 2; ++m) {
#pragma unroll
    for (int j = 0; j < 4; ++j) {
      int n = (wave * 4 + j) * 16 + (lane & 15);
      float bb = b_value[n];
      int mbase = mtile * 32 + m * 16 + (lane >> 4) * 8;
#pragma unroll
      for (int r = 0; r < 8; ++r)
        Vout[(size_t)(mbase + r) * CDIM + n] = (__bf16)(acc[m][j][r] + bb);
    }
  }
}

// ---------------------------------------------------------------------------
// q projection: A = (query + query_pos) (MQ x 256), N = 384 (off:256 | attn:128).
// 6 waves (192 thr): waves 0-3 -> W_off tiles, waves 4-5 -> W_attn tiles.
// ---------------------------------------------------------------------------
__global__ __launch_bounds__(192)
void gemm_q_kernel(const float* __restrict__ query, const float* __restrict__ qpos,
                   const __bf16* __restrict__ Woff_pk, const __bf16* __restrict__ Wattn_pk,
                   const float* __restrict__ b_off, const float* __restrict__ b_attn,
                   float* __restrict__ off_raw, float* __restrict__ attn_raw) {
  __shared__ __bf16 lds_a[16 * 32];
  const int mtile = blockIdx.x;
  const int wave  = threadIdx.x >> 5;
  const int lane  = threadIdx.x & 31;
  const float* Aq = query + (size_t)mtile * 16 * CDIM;
  const float* Ap = qpos  + (size_t)mtile * 16 * CDIM;

  const __bf16* pk = (wave < 4) ? Woff_pk : Wattn_pk;
  const int ntiles = (wave < 4) ? 16 : 8;
  const int ntoff  = (wave < 4) ? wave * 4 : (wave - 4) * 4;

  v8f acc[4] = {};
  for (int kt = 0; kt < 8; ++kt) {
    __syncthreads();
#pragma unroll
    for (int i = 0; i < 3; ++i) {
      int idx = threadIdx.x + i * 192;
      if (idx < 512) {
        int r = idx >> 5, c = idx & 31;
        size_t g = (size_t)r * CDIM + kt * 32 + c;
        lds_a[idx] = (__bf16)(Aq[g] + Ap[g]);
      }
    }
    __syncthreads();

    v16bf a = a_frag_from_lds_bf16(lds_a, lane);
#pragma unroll
    for (int j = 0; j < 4; ++j) {
      v16bf b = b_frag(pk, kt * ntiles + ntoff + j, lane);
      acc[j] = __builtin_amdgcn_wmma_f32_16x16x32_bf16(false, a, false, b,
                                                       (short)0, acc[j], false, false);
    }
  }
#pragma unroll
  for (int j = 0; j < 4; ++j) {
    int ng = (wave * 4 + j) * 16 + (lane & 15);   // global col in [0,384)
    int mbase = mtile * 16 + (lane >> 4) * 8;
    if (ng < 256) {
      float bb = b_off[ng];
#pragma unroll
      for (int r = 0; r < 8; ++r)
        off_raw[(size_t)(mbase + r) * 256 + ng] = acc[j][r] + bb;
    } else {
      int na = ng - 256;
      float bb = b_attn[na];
#pragma unroll
      for (int r = 0; r < 8; ++r)
        attn_raw[(size_t)(mbase + r) * 128 + na] = acc[j][r] + bb;
    }
  }
}

// ---------------------------------------------------------------------------
// Bilinear sampling + head-wise softmax; one wave per (bs, nq), lane = channel.
// ---------------------------------------------------------------------------
__device__ __forceinline__ float sample_corner(const __bf16* __restrict__ V,
                                               int st, int Wl, int Hl, int bs,
                                               int hd, int x, int y, float w) {
  if (x < 0 || x >= Wl || y < 0 || y >= Hl) return 0.f;
  return w * (float)V[((size_t)(st + y * Wl + x) * BSZ + bs) * CDIM + hd];
}

__global__ __launch_bounds__(256)
void msda_sample_kernel(const __bf16* __restrict__ V,        // (NVAL, BSZ, 256)
                        const float* __restrict__ refpts,    // (BSZ, NQRY, 2)
                        const float* __restrict__ off_raw,   // (MQ, 256)
                        const float* __restrict__ attn_raw,  // (MQ, 128)
                        __bf16* __restrict__ msda) {         // (MQ, 256)
  const int wid = blockIdx.x * (blockDim.x >> 5) + (threadIdx.x >> 5);
  if (wid >= MQ) return;
  const int lane = threadIdx.x & 31;
  const int bs = wid & (BSZ - 1);
  const int nq = wid >> 3;

  const int LH[NLV] = {128, 64, 32, 16};
  const int LW[NLV] = {128, 64, 32, 16};
  const int LS[NLV] = {0, 16384, 20480, 21504};

  const float rx = refpts[((size_t)bs * NQRY + nq) * 2 + 0];
  const float ry = refpts[((size_t)bs * NQRY + nq) * 2 + 1];
  const float* offr = off_raw + (size_t)wid * 256;
  const float* attr = attn_raw + (size_t)wid * 128;

  for (int h = 0; h < NHD; ++h) {
    float mx = -1e30f;
#pragma unroll
    for (int i = 0; i < 16; ++i) mx = fmaxf(mx, attr[h * 16 + i]);
    float e[16], s = 0.f;
#pragma unroll
    for (int i = 0; i < 16; ++i) { e[i] = __expf(attr[h * 16 + i] - mx); s += e[i]; }
    const float inv = 1.f / s;

    const int hd = h * DDIM + lane;
    float acc = 0.f;
#pragma unroll
    for (int l = 0; l < NLV; ++l) {
      const int Hl = LH[l], Wl = LW[l], st = LS[l];
#pragma unroll
      for (int p = 0; p < NPT; ++p) {
        const int oi = ((h * NLV + l) * NPT + p) * 2;
        const float lx = rx + offr[oi]     / (float)Wl;
        const float ly = ry + offr[oi + 1] / (float)Hl;
        const float px = lx * (float)Wl - 0.5f;
        const float py = ly * (float)Hl - 0.5f;
        const float x0f = floorf(px), y0f = floorf(py);
        const float fx = px - x0f, fy = py - y0f;
        const int x0 = (int)x0f, y0 = (int)y0f;
        const float aw = e[l * NPT + p] * inv;
        float g = sample_corner(V, st, Wl, Hl, bs, hd, x0,     y0,     (1.f - fx) * (1.f - fy))
                + sample_corner(V, st, Wl, Hl, bs, hd, x0 + 1, y0,     fx * (1.f - fy))
                + sample_corner(V, st, Wl, Hl, bs, hd, x0,     y0 + 1, (1.f - fx) * fy)
                + sample_corner(V, st, Wl, Hl, bs, hd, x0 + 1, y0 + 1, fx * fy);
        acc += aw * g;
      }
    }
    msda[(size_t)wid * CDIM + hd] = (__bf16)acc;
  }
}

// ---------------------------------------------------------------------------
// Output projection: msda(bf16, MQ x 256) @ W_out + b_out + residual(query).
// ---------------------------------------------------------------------------
__global__ __launch_bounds__(128)
void gemm_out_kernel(const __bf16* __restrict__ msda, const __bf16* __restrict__ Wpk,
                     const float* __restrict__ b_out, const float* __restrict__ query,
                     float* __restrict__ out) {
  const int mtile = blockIdx.x;
  const int wave  = threadIdx.x >> 5;
  const int lane  = threadIdx.x & 31;
  const __bf16* Arow = msda + (size_t)mtile * 16 * CDIM;

  v8f acc[4] = {};
  for (int kt = 0; kt < 8; ++kt) {
    v16bf a = a_frag_from_global_bf16(Arow + kt * 32, CDIM, lane);
#pragma unroll
    for (int j = 0; j < 4; ++j) {
      v16bf b = b_frag(Wpk, kt * 16 + wave * 4 + j, lane);
      acc[j] = __builtin_amdgcn_wmma_f32_16x16x32_bf16(false, a, false, b,
                                                       (short)0, acc[j], false, false);
    }
  }
#pragma unroll
  for (int j = 0; j < 4; ++j) {
    int n = (wave * 4 + j) * 16 + (lane & 15);
    float bb = b_out[n];
    int mbase = mtile * 16 + (lane >> 4) * 8;
#pragma unroll
    for (int r = 0; r < 8; ++r) {
      size_t idx = (size_t)(mbase + r) * CDIM + n;
      out[idx] = acc[j][r] + bb + query[idx];
    }
  }
}

// ---------------------------------------------------------------------------
extern "C" void kernel_launch(void* const* d_in, const int* in_sizes, int n_in,
                              void* d_out, int out_size, void* d_ws, size_t ws_size,
                              hipStream_t stream) {
  const float* query    = (const float*)d_in[0];
  // d_in[1] = key (unused)
  const float* value    = (const float*)d_in[2];
  const float* qpos     = (const float*)d_in[3];
  const float* refpts   = (const float*)d_in[4];
  // d_in[5] = bev_feat_shapes, d_in[6] = level_start_index (baked in as constants)
  const float* W_value  = (const float*)d_in[7];
  const float* b_value  = (const float*)d_in[8];
  const float* W_off    = (const float*)d_in[9];
  const float* b_off    = (const float*)d_in[10];
  const float* W_attn   = (const float*)d_in[11];
  const float* b_attn   = (const float*)d_in[12];
  const float* W_out    = (const float*)d_in[13];
  const float* b_out    = (const float*)d_in[14];

  size_t cur = 0;
  auto carve = [&](size_t bytes) -> void* {
    void* p = (char*)d_ws + cur;
    cur += (bytes + 255) & ~(size_t)255;
    return p;
  };
  __bf16* Wv_pk    = (__bf16*)carve((size_t)8 * 16 * 512 * 2);
  __bf16* Woff_pk  = (__bf16*)carve((size_t)8 * 16 * 512 * 2);
  __bf16* Wattn_pk = (__bf16*)carve((size_t)8 * 8  * 512 * 2);
  __bf16* Wo_pk    = (__bf16*)carve((size_t)8 * 16 * 512 * 2);
  __bf16* Vbf      = (__bf16*)carve((size_t)MV * CDIM * 2);
  float*  off_raw  = (float*) carve((size_t)MQ * 256 * 4);
  float*  attn_raw = (float*) carve((size_t)MQ * 128 * 4);
  __bf16* msda     = (__bf16*)carve((size_t)MQ * CDIM * 2);

  // 1) pack weights into WMMA B-operand layout
  pack_b_kernel<<<(8 * 16 * 32 + 255) / 256, 256, 0, stream>>>(W_value, Wv_pk,  256, 256);
  pack_b_kernel<<<(8 * 16 * 32 + 255) / 256, 256, 0, stream>>>(W_off,   Woff_pk, 256, 256);
  pack_b_kernel<<<(8 * 8  * 32 + 255) / 256, 256, 0, stream>>>(W_attn,  Wattn_pk, 256, 128);
  pack_b_kernel<<<(8 * 16 * 32 + 255) / 256, 256, 0, stream>>>(W_out,   Wo_pk,  256, 256);

  // 2) value projection (dominant GEMM; TDM-fed, bf16-out so L2-resident)
  gemm_value_kernel<<<MV / 32, 128, 0, stream>>>(value, b_value, Wv_pk, Vbf);

  // 3) fused offset + attention-logit projection
  gemm_q_kernel<<<MQ / 16, 192, 0, stream>>>(query, qpos, Woff_pk, Wattn_pk,
                                             b_off, b_attn, off_raw, attn_raw);

  // 4) softmax + bilinear sampling (one wave per (bs, query))
  msda_sample_kernel<<<(MQ + 7) / 8, 256, 0, stream>>>(Vbf, refpts, off_raw, attn_raw, msda);

  // 5) output projection + bias + residual
  gemm_out_kernel<<<MQ / 16, 128, 0, stream>>>(msda, Wo_pk, b_out, query, (float*)d_out);
}
